// NKFStreaming_81664508166196
// MI455X (gfx1250) — matrix operand order repaired and checked by reading, not compiled, not run on hardware
//
#include <hip/hip_runtime.h>
#include <hip/hip_bf16.h>
#include <math.h>

typedef __bf16 bf16_t;
typedef __attribute__((ext_vector_type(16))) __bf16 v16bf;
typedef __attribute__((ext_vector_type(8)))  float  v8f;

#define NCH   128
#define HOP   256
#define BLK   1024
#define NF    513
#define NTAP  4
#define NH    64
#define NTIL  33     // ceil(513/16) bin tiles
#define FPAD  528    // padded bin count (33*16)

// ---- workspace layout (float offsets into d_ws) ----
#define O_COS 0          // 1024 cos table
#define O_FX  1024       // 1024 input frame buffer x
#define O_FY  2048       // 1024 input frame buffer y
#define O_WFX 3072       // 1024 windowed frame x
#define O_WFY 4096       // 1024 windowed frame y
#define O_XF  5120       // 513*2 Xf (re,im)    (reserve 1056)
#define O_YF  6176       // 513*2 Yf
#define O_E   7232       // 513*2 E = Yf - echo
#define O_XH  8288       // 528*4*2 X_hist
#define O_HP  12512      // 528*4*2 h_prior
#define O_HQ  16736      // 528*4*2 h_post
#define O_OB  20960      // 1024 overlap-add buffer
#define O_PP  21984      // 128*9 pred partials
#define O_PF  23136      // 128 pred flags
#define O_GH  23264      // 4 chains * 2 layers * 528 * 64 GRU hidden
#define F32_TOT 293600
#define WB16_BYTE (F32_TOT*4)   // start of bf16 swizzled weights (byte offset)
#define NWELEM 98304            // 8 matrices * 12288 bf16 elements

// A-matrix (16x32, bf16) per-lane element index for (row m, k) -> linear offset
__device__ __forceinline__ int aoff(int m, int k) {
  int half, j;
  if (k < 16) { half = k >> 3; j = k & 7; }
  else { int kk = k - 16; half = kk >> 3; j = 8 + (kk & 7); }
  return ((m + 16 * half) << 4) + j;
}

__device__ __forceinline__ v8f wmma_bf16(v16bf a, v16bf b, v8f c) {
  return __builtin_amdgcn_wmma_f32_16x16x32_bf16(false, a, false, b, (short)0, c, false, false);
}

// hardware-rate sigmoid/tanh: v_exp_f32 + v_rcp_f32 (TRANS ops co-execute with WMMA)
__device__ __forceinline__ float sigm(float x) {
  return __builtin_amdgcn_rcpf(1.f + __expf(-x));
}
__device__ __forceinline__ float tanh_fast(float x) {
  return 2.f * __builtin_amdgcn_rcpf(1.f + __expf(-2.f * x)) - 1.f;
}

// ---------------- init: cos table, zero state, swizzle weights to bf16 B-layout ----------------
__global__ __launch_bounds__(256) void k_init(float* ws, bf16_t* wsw,
    const float* rwih, const float* rwhh, const float* iwih, const float* iwhh) {
  int idx = blockIdx.x * 256 + threadIdx.x;
  if (idx < F32_TOT) {
    float v = 0.f;
    if (idx < 1024) v = cosf(6.283185307179586f * (float)idx / 1024.f);
    ws[idx] = v;
  }
  if (idx < NWELEM) {
    int mat = idx / 12288, r = idx % 12288;
    int tile = r >> 9, e = r & 511;
    int lane = e >> 4, j = e & 15;
    int nt = tile >> 1, kb = tile & 1;
    int n = nt * 16 + (lane & 15);
    int k = kb * 32 + ((j >> 3) << 4) + ((lane >> 4) << 3) + (j & 7);
    int set = mat >> 2, type = (mat >> 1) & 1, layer = mat & 1;
    const float* src = set ? (type ? iwhh : iwih) : (type ? rwhh : rwih);
    wsw[idx] = (bf16_t)src[layer * 12288 + n * 64 + k];
  }
}

// ---------------- per-step: shift frame buffers + window ----------------
__global__ __launch_bounds__(256) void k_frame(float* ws, const float* x, const float* y,
                                               const float* window, int s) {
  __shared__ float ox[1024], oy[1024];
  int t = threadIdx.x;
  for (int i = t; i < 1024; i += 256) { ox[i] = ws[O_FX + i]; oy[i] = ws[O_FY + i]; }
  __syncthreads();
  for (int i = t; i < 1024; i += 256) {
    float nx = (i < 768) ? ox[i + 256] : x[s * HOP + (i - 768)];
    float ny = (i < 768) ? oy[i + 256] : y[s * HOP + (i - 768)];
    ws[O_FX + i] = nx; ws[O_FY + i] = ny;
    float w = window[i];
    ws[O_WFX + i] = nx * w; ws[O_WFY + i] = ny * w;
  }
}

// ---------------- per-step: table-driven rDFT (both frames) + pred partial sums ----------------
__global__ __launch_bounds__(128) void k_dft(float* ws, int s) {
  __shared__ float wf[2][1024];
  __shared__ float ct[1024];
  __shared__ float red[128];
  int t = threadIdx.x;
  for (int i = t; i < 1024; i += 128) {
    wf[0][i] = ws[O_WFX + i];
    wf[1][i] = ws[O_WFY + i];
    ct[i] = ws[O_COS + i];
  }
  __syncthreads();
  int task = blockIdx.x * 128 + t;
  float contrib = 0.f;
  if (task < 2 * NF) {
    int frame = task >= NF;
    int k = task - frame * NF;
    float sr = 0.f, si = 0.f;
    for (int n2 = 0; n2 < 1024; ++n2) {
      int p = (k * n2) & 1023;
      float v = wf[frame][n2];
      sr += v * ct[p];
      si -= v * ct[(p + 768) & 1023];
    }
    int ob = frame ? O_YF : O_XF;
    ws[ob + k * 2] = sr; ws[ob + k * 2 + 1] = si;
    if (!frame) {
      contrib = sqrtf(sr * sr + si * si);
      for (int tp = 0; tp < 3; ++tp) {
        float a = ws[O_XH + k * 8 + tp * 2], b = ws[O_XH + k * 8 + tp * 2 + 1];
        contrib += sqrtf(a * a + b * b);
      }
    }
  }
  red[t] = contrib;
  __syncthreads();
  for (int off = 64; off > 0; off >>= 1) {
    if (t < off) red[t] += red[t + off];
    __syncthreads();
  }
  if (t == 0) ws[O_PP + s * 9 + blockIdx.x] = red[0];
}

// ---------------- per-step: combine pred partials ----------------
__global__ void k_pred(float* ws, int s) {
  if (threadIdx.x == 0 && blockIdx.x == 0) {
    float sum = 0.f;
    for (int i = 0; i < 9; ++i) sum += ws[O_PP + s * 9 + i];
    ws[O_PF + s] = (sum / (float)(NF * NTAP) >= 1e-5f) ? 1.f : 0.f;
  }
}

// ---------------- per-step: the heavy per-bin update kernel (WMMA GRUs) ----------------
__global__ __launch_bounds__(128) void k_upd(float* ws, const bf16_t* wsw,
    const float* fiwr, const float* fiwi, const float* fibr, const float* fibi,
    const float* rbih, const float* rbhh, const float* ibih, const float* ibhh,
    const float* fowr, const float* fowi, const float* fobr, const float* fobi,
    int s) {
  __shared__ float sXhR[16][4], sXhI[16][4];
  __shared__ float sEr[16], sEi[16];
  __shared__ float sHpR[16][4], sHpI[16][4];
  __shared__ float featR[16][9], featI[16][9];
  __shared__ alignas(32) bf16_t frA[2][512];
  __shared__ alignas(32) bf16_t fiA[2][512];
  __shared__ alignas(32) bf16_t hA[4][2][512];
  __shared__ alignas(32) bf16_t xA[4][2][512];
  __shared__ float Fout[4][16][64];

  const int t = threadIdx.x;
  const int b0 = blockIdx.x * 16;
  const int wave = t >> 5, lane = t & 31;
  const float pred = ws[O_PF + s];

  // P0: per-bin scalar state: shift X_hist, compute e, dh, hp2, features
  if (t < 16) {
    int bin = b0 + t;
    bool ok = bin < NF;
    float xr[4], xi[4], pr[4], pi[4], qr[4], qi[4], yr = 0.f, yi = 0.f;
    if (ok) {
      xr[0] = ws[O_XF + bin * 2]; xi[0] = ws[O_XF + bin * 2 + 1];
      for (int tp = 1; tp < 4; ++tp) {
        xr[tp] = ws[O_XH + bin * 8 + (tp - 1) * 2];
        xi[tp] = ws[O_XH + bin * 8 + (tp - 1) * 2 + 1];
      }
      for (int tp = 0; tp < 4; ++tp) {
        ws[O_XH + bin * 8 + tp * 2]     = xr[tp];
        ws[O_XH + bin * 8 + tp * 2 + 1] = xi[tp];
        pr[tp] = ws[O_HP + bin * 8 + tp * 2]; pi[tp] = ws[O_HP + bin * 8 + tp * 2 + 1];
        qr[tp] = ws[O_HQ + bin * 8 + tp * 2]; qi[tp] = ws[O_HQ + bin * 8 + tp * 2 + 1];
      }
      yr = ws[O_YF + bin * 2]; yi = ws[O_YF + bin * 2 + 1];
    } else {
      for (int tp = 0; tp < 4; ++tp) { xr[tp]=xi[tp]=pr[tp]=pi[tp]=qr[tp]=qi[tp]=0.f; }
    }
    float er = yr, ei = yi;
    for (int tp = 0; tp < 4; ++tp) {
      er -= xr[tp] * qr[tp] - xi[tp] * qi[tp];
      ei -= xr[tp] * qi[tp] + xi[tp] * qr[tp];
      sXhR[t][tp] = xr[tp]; sXhI[t][tp] = xi[tp];
      sHpR[t][tp] = qr[tp]; sHpI[t][tp] = qi[tp];
      featR[t][tp] = xr[tp]; featI[t][tp] = xi[tp];
      featR[t][5 + tp] = qr[tp] - pr[tp]; featI[t][5 + tp] = qi[tp] - pi[tp];
    }
    sEr[t] = er; sEi[t] = ei;
    featR[t][4] = er; featI[t][4] = ei;
  }
  __syncthreads();

  if (pred != 0.f) {   // uniform across grid
    // P1: fc_in (complex 9 -> 64), write fr/fi straight into WMMA A-layout (bf16)
    {
      int m = t >> 3, h0 = (t & 7) * 8;
      for (int u = 0; u < 8; ++u) {
        int h = h0 + u;
        float fr = fibr[h], fi = fibi[h];
        for (int f = 0; f < 9; ++f) {
          float wr = fiwr[h * 9 + f], wi = fiwi[h * 9 + f];
          float ar = featR[m][f], ai = featI[m][f];
          fr += ar * wr - ai * wi;
          fi += ar * wi + ai * wr;
        }
        int kb = h >> 5, o = aoff(m, h & 31);
        frA[kb][o] = (bf16_t)fr;
        fiA[kb][o] = (bf16_t)fi;
      }
    }
    __syncthreads();

    // P2: one GRU chain per wave: 0=Frr(r,fr) 1=Fir(i,fr) 2=Fri(r,fi) 3=Fii(i,fi)
    const int set  = wave & 1;
    const int xsel = wave >> 1;
    const float* bih_all = set ? ibih : rbih;
    const float* bhh_all = set ? ibhh : rbhh;
    for (int l = 0; l < 2; ++l) {
      float* hst = ws + O_GH + (size_t)((wave * 2 + l) * FPAD + b0) * 64;
      for (int idx = lane; idx < 1024; idx += 32) {
        int m = idx >> 6, k = idx & 63;
        hA[wave][k >> 5][aoff(m, k & 31)] = (bf16_t)hst[m * 64 + k];
      }
      const bf16_t* xsrc0; const bf16_t* xsrc1;
      if (l == 0) { xsrc0 = xsel ? &fiA[0][0] : &frA[0][0]; xsrc1 = xsel ? &fiA[1][0] : &frA[1][0]; }
      else        { xsrc0 = &xA[wave][0][0];                xsrc1 = &xA[wave][1][0]; }
      v16bf ax0 = *(const v16bf*)(xsrc0 + lane * 16);
      v16bf ax1 = *(const v16bf*)(xsrc1 + lane * 16);
      v16bf ah0 = *(const v16bf*)(&hA[wave][0][0] + lane * 16);
      v16bf ah1 = *(const v16bf*)(&hA[wave][1][0] + lane * 16);
      const bf16_t* Wih = wsw + (size_t)(((set * 2 + 0) * 2) + l) * 12288;
      const bf16_t* Whh = wsw + (size_t)(((set * 2 + 1) * 2) + l) * 12288;
      const float* bih = bih_all + l * 192;
      const float* bhh = bhh_all + l * 192;
      for (int jt = 0; jt < 4; ++jt) {
        int n = lane & 15;
        float br_ = bih[jt * 16 + n] + bhh[jt * 16 + n];
        float bz_ = bih[64 + jt * 16 + n] + bhh[64 + jt * 16 + n];
        float bi_ = bih[128 + jt * 16 + n];
        float bh_ = bhh[128 + jt * 16 + n];
        v8f Sr, Sz, In, Hn;
        for (int r = 0; r < 8; ++r) { Sr[r] = br_; Sz[r] = bz_; In[r] = bi_; Hn[r] = bh_; }
        Sr = wmma_bf16(ax0, *(const v16bf*)(Wih + (jt * 2 + 0) * 512 + lane * 16), Sr);
        Sr = wmma_bf16(ax1, *(const v16bf*)(Wih + (jt * 2 + 1) * 512 + lane * 16), Sr);
        Sr = wmma_bf16(ah0, *(const v16bf*)(Whh + (jt * 2 + 0) * 512 + lane * 16), Sr);
        Sr = wmma_bf16(ah1, *(const v16bf*)(Whh + (jt * 2 + 1) * 512 + lane * 16), Sr);
        Sz = wmma_bf16(ax0, *(const v16bf*)(Wih + ((4 + jt) * 2 + 0) * 512 + lane * 16), Sz);
        Sz = wmma_bf16(ax1, *(const v16bf*)(Wih + ((4 + jt) * 2 + 1) * 512 + lane * 16), Sz);
        Sz = wmma_bf16(ah0, *(const v16bf*)(Whh + ((4 + jt) * 2 + 0) * 512 + lane * 16), Sz);
        Sz = wmma_bf16(ah1, *(const v16bf*)(Whh + ((4 + jt) * 2 + 1) * 512 + lane * 16), Sz);
        In = wmma_bf16(ax0, *(const v16bf*)(Wih + ((8 + jt) * 2 + 0) * 512 + lane * 16), In);
        In = wmma_bf16(ax1, *(const v16bf*)(Wih + ((8 + jt) * 2 + 1) * 512 + lane * 16), In);
        Hn = wmma_bf16(ah0, *(const v16bf*)(Whh + ((8 + jt) * 2 + 0) * 512 + lane * 16), Hn);
        Hn = wmma_bf16(ah1, *(const v16bf*)(Whh + ((8 + jt) * 2 + 1) * 512 + lane * 16), Hn);
        // gate nonlinearity directly on accumulators: element (r) = row m, col j
        int j = jt * 16 + n;
        for (int r = 0; r < 8; ++r) {
          int m = r + ((lane >> 4) << 3);
          float hold = hst[m * 64 + j];
          float rg = sigm(Sr[r]);
          float zg = sigm(Sz[r]);
          float ng = tanh_fast(In[r] + rg * Hn[r]);
          float hn = (1.f - zg) * ng + zg * hold;
          hst[m * 64 + j] = hn;
          if (l == 0) xA[wave][j >> 5][aoff(m, j & 31)] = (bf16_t)hn;
          else        Fout[wave][m][j] = hn;
        }
      }
    }
    __syncthreads();

    // P3: fc_out (64 -> 4 complex) + Kalman gain update per (bin, tap)
    if (t < 64) {
      int m = t >> 2, li = t & 3;
      int bin = b0 + m;
      float kr = fobr[li], ki = fobi[li];
      for (int h = 0; h < 64; ++h) {
        float oR = Fout[0][m][h] - Fout[3][m][h];
        float oI = Fout[2][m][h] + Fout[1][m][h];
        float wr = fowr[li * 64 + h], wi = fowi[li * 64 + h];
        kr += oR * wr - oI * wi;
        ki += oR * wi + oI * wr;
      }
      if (bin < NF) {
        float er = sEr[m], ei = sEi[m];
        float hr = sHpR[m][li] + (kr * er - ki * ei);
        float hi = sHpI[m][li] + (kr * ei + ki * er);
        ws[O_HP + bin * 8 + li * 2]     = sHpR[m][li];
        ws[O_HP + bin * 8 + li * 2 + 1] = sHpI[m][li];
        ws[O_HQ + bin * 8 + li * 2]     = hr;
        ws[O_HQ + bin * 8 + li * 2 + 1] = hi;
      }
    }
    __syncthreads();
  }

  // echo with (possibly updated) h_post, E = Yf - echo
  if (t < 16) {
    int bin = b0 + t;
    if (bin < NF) {
      float er = ws[O_YF + bin * 2], ei = ws[O_YF + bin * 2 + 1];
      for (int tp = 0; tp < 4; ++tp) {
        float hr = ws[O_HQ + bin * 8 + tp * 2];
        float hi = ws[O_HQ + bin * 8 + tp * 2 + 1];
        er -= sXhR[t][tp] * hr - sXhI[t][tp] * hi;
        ei -= sXhR[t][tp] * hi + sXhI[t][tp] * hr;
      }
      ws[O_E + bin * 2]     = er;
      ws[O_E + bin * 2 + 1] = ei;
    }
  }
}

// ---------------- per-step: irDFT + window + overlap-add + emit chunk ----------------
__global__ __launch_bounds__(256) void k_synth(float* ws, const float* window, float* out, int s) {
  __shared__ float er[513], ei[513];
  __shared__ float ct[1024], sf[1024], ob[1024];
  int t = threadIdx.x;
  for (int i = t; i < 1024; i += 256) { ct[i] = ws[O_COS + i]; ob[i] = ws[O_OB + i]; }
  for (int i = t; i < NF; i += 256)   { er[i] = ws[O_E + i * 2]; ei[i] = ws[O_E + i * 2 + 1]; }
  __syncthreads();
  for (int n2 = t; n2 < 1024; n2 += 256) {
    float acc = er[0] + ((n2 & 1) ? -er[512] : er[512]);
    for (int k = 1; k < 512; ++k) {
      int p = (k * n2) & 1023;
      acc += 2.f * (er[k] * ct[p] - ei[k] * ct[(p + 768) & 1023]);
    }
    sf[n2] = acc * (1.f / 1024.f) * window[n2];
  }
  __syncthreads();
  for (int i = t; i < 1024; i += 256) {
    float v = ob[i] + sf[i];
    if (i < 256) out[s * HOP + i] = v;
    ws[O_OB + i] = (i < 768) ? (ob[i + 256] + sf[i + 256]) : 0.f;
  }
}

extern "C" void kernel_launch(void* const* d_in, const int* in_sizes, int n_in,
                              void* d_out, int out_size, void* d_ws, size_t ws_size,
                              hipStream_t stream) {
  (void)in_sizes; (void)n_in; (void)out_size; (void)ws_size;
  const float* x        = (const float*)d_in[0];
  const float* y        = (const float*)d_in[1];
  const float* window   = (const float*)d_in[2];
  const float* fiwr     = (const float*)d_in[3];
  const float* fiwi     = (const float*)d_in[4];
  const float* fibr     = (const float*)d_in[5];
  const float* fibi     = (const float*)d_in[6];
  const float* rwih     = (const float*)d_in[7];
  const float* rwhh     = (const float*)d_in[8];
  const float* rbih     = (const float*)d_in[9];
  const float* rbhh     = (const float*)d_in[10];
  const float* iwih     = (const float*)d_in[11];
  const float* iwhh     = (const float*)d_in[12];
  const float* ibih     = (const float*)d_in[13];
  const float* ibhh     = (const float*)d_in[14];
  const float* fowr     = (const float*)d_in[15];
  const float* fowi     = (const float*)d_in[16];
  const float* fobr     = (const float*)d_in[17];
  const float* fobi     = (const float*)d_in[18];

  float*  wsf = (float*)d_ws;
  bf16_t* wsw = (bf16_t*)((char*)d_ws + WB16_BYTE);
  float*  out = (float*)d_out;

  k_init<<<(F32_TOT + 255) / 256, 256, 0, stream>>>(wsf, wsw, rwih, rwhh, iwih, iwhh);
  for (int s = 0; s < NCH; ++s) {
    k_frame<<<1, 256, 0, stream>>>(wsf, x, y, window, s);
    k_dft<<<9, 128, 0, stream>>>(wsf, s);
    k_pred<<<1, 32, 0, stream>>>(wsf, s);
    k_upd<<<NTIL, 128, 0, stream>>>(wsf, wsw, fiwr, fiwi, fibr, fibi,
                                    rbih, rbhh, ibih, ibhh,
                                    fowr, fowi, fobr, fobi, s);
    k_synth<<<1, 256, 0, stream>>>(wsf, window, out, s);
  }
}
